// Graphcnn_geo_3513283248956
// MI455X (gfx1250) — matrix operand order repaired and checked by reading, not compiled, not run on hardware
//
#include <hip/hip_runtime.h>
#include <hip/hip_bf16.h>

#ifndef __has_builtin
#define __has_builtin(x) 0
#endif

typedef _Float16 v16h __attribute__((ext_vector_type(16)));
typedef float    v8f  __attribute__((ext_vector_type(8)));
typedef float    v2f  __attribute__((ext_vector_type(2)));

#define DEVFN __device__ __forceinline__

constexpr int   BB    = 4;
constexpr int   NN    = 2048;
constexpr int   KNB   = 20;          // neighbors
constexpr int   DROP  = 6;           // 20 - ceil(20*2/3)
constexpr float EPSB  = 1e-5f;
constexpr float SLOPE = 0.2f;
constexpr int   NT    = NN / 16;     // 128 16-wide tiles
constexpr int   JT    = 4;           // j-tiles per wave in dist GEMM
constexpr int   LDSW  = 520;         // edgeconv: max Kpad(512) + 8 halfs pad
constexpr int   S9    = 20;          // gemm9 LDS stride in floats (80B, 16B-aligned rows)

#if __has_builtin(__builtin_amdgcn_global_load_async_to_lds_b128)
#define ATHENA_ASYNC_LDS 1
typedef int v4i_t __attribute__((vector_size(16)));
typedef __attribute__((address_space(1))) v4i_t GV4;   // global int4*
typedef __attribute__((address_space(3))) v4i_t LV4;   // LDS int4*
#else
#define ATHENA_ASYNC_LDS 0
#endif

DEVFN float lrelu(float v) { return v > 0.f ? v : SLOPE * v; }

// ---------------------------------------------------------------- pad x [B,3,N] -> [B,4,N] (4th channel zero)
__global__ void pad_kernel(const float* __restrict__ x, float* __restrict__ xp) {
  int i = blockIdx.x * 256 + threadIdx.x;
  if (i >= BB * 4 * NN) return;
  int n = i % NN, c = (i / NN) % 4, b = i / (4 * NN);
  xp[i] = (c < 3) ? x[((size_t)b * 3 + c) * NN + n] : 0.f;
}

// ---------------------------------------------------------------- xx = sum_c x^2
__global__ void xx_kernel(const float* __restrict__ F, int CS, int co, int C,
                          float* __restrict__ xx) {
  int i = blockIdx.x * 256 + threadIdx.x;
  if (i >= BB * NN) return;
  int n = i % NN, b = i / NN;
  const float* Fb = F + ((size_t)b * CS + co) * NN + n;
  float s = 0.f;
  for (int c = 0; c < C; ++c) { float v = Fb[(size_t)c * NN]; s += v * v; }
  xx[i] = s;
}

// ---------------------------------------------------------------- pd = 2 X^T X - xx_i - xx_j
// one wave per 16x64 strip (4 j-tiles), f32 WMMA 16x16x4, C % 4 == 0 (no guards)
__global__ void dist_kernel(const float* __restrict__ F, int CS, int co, int C,
                            const float* __restrict__ xx, float* __restrict__ pd) {
  int wave = threadIdx.x >> 5;
  int lane = threadIdx.x & 31;
  int hi   = lane >> 4, l16 = lane & 15;
  int tile = blockIdx.x * 8 + wave;
  int tpb  = NT * (NT / JT);
  int b    = tile / tpb;
  int r    = tile % tpb;
  int i0   = (r / (NT / JT)) * 16;
  int j0   = (r % (NT / JT)) * (16 * JT);

  const float* Fb  = F + ((size_t)b * CS + co) * NN;
  const float* FbA = Fb + i0 + l16;
  const float* FbB = Fb + j0 + l16;

  v8f acc[JT];
#pragma unroll
  for (int t = 0; t < JT; ++t) acc[t] = (v8f){};

  for (int k0 = 0; k0 < C; k0 += 4) {
    size_t ro0 = (size_t)(k0 + 2 * hi) * NN;
    size_t ro1 = ro0 + NN;
    v2f A;
    A[0] = FbA[ro0];
    A[1] = FbA[ro1];
#pragma unroll
    for (int t = 0; t < JT; ++t) {
      v2f Bv;
      Bv[0] = FbB[ro0 + 16 * t];
      Bv[1] = FbB[ro1 + 16 * t];
      acc[t] = __builtin_amdgcn_wmma_f32_16x16x4_f32(false, A, false, Bv,
                                                     (short)0, acc[t], false, false);
    }
  }

  const float* xb = xx + (size_t)b * NN;
  float xi[8];
#pragma unroll
  for (int v = 0; v < 8; ++v) xi[v] = xb[i0 + v + 8 * hi];
#pragma unroll
  for (int t = 0; t < JT; ++t) {
    float xj = xb[j0 + 16 * t + l16];
#pragma unroll
    for (int v = 0; v < 8; ++v) {
      int m = v + 8 * hi;
      pd[((size_t)b * NN + i0 + m) * NN + j0 + 16 * t + l16] =
          2.f * acc[t][v] - xi[v] - xj;
    }
  }
}

// ---------------------------------------------------------------- masked top-k (k=20) per row
__global__ void topk_kernel(const float* __restrict__ pd, const int* __restrict__ lidx,
                            int invert, int* __restrict__ idx) {
  __shared__ float sv[NN];
  __shared__ float rv[256];
  __shared__ int   ri[256];
  int tid = threadIdx.x;
  int b   = blockIdx.x / NN;
  const float* rowp = pd + (size_t)blockIdx.x * NN;

  float lmin = 3.4e38f;
  for (int j = tid; j < NN; j += 256) { float v = rowp[j]; sv[j] = v; lmin = fminf(lmin, v); }
  rv[tid] = lmin; __syncthreads();
  for (int s = 128; s > 0; s >>= 1) {
    if (tid < s) rv[tid] = fminf(rv[tid], rv[tid + s]);
    __syncthreads();
  }
  float mn = rv[0]; __syncthreads();

  for (int j = tid; j < NN; j += 256) {
    int keep = ((lidx[(size_t)b * NN + j] != 0) ? 1 : 0) ^ invert;
    if (!keep) sv[j] = mn;
  }
  __syncthreads();

  int* op = idx + (size_t)blockIdx.x * KNB;
  for (int t = 0; t < KNB; ++t) {
    float bv = -3.4e38f; int bi = NN;
    for (int j = tid; j < NN; j += 256) {
      float v = sv[j];
      if (v > bv || (v == bv && j < bi)) { bv = v; bi = j; }
    }
    rv[tid] = bv; ri[tid] = bi; __syncthreads();
    for (int s = 128; s > 0; s >>= 1) {
      if (tid < s) {
        if (rv[tid + s] > rv[tid] || (rv[tid + s] == rv[tid] && ri[tid + s] < ri[tid])) {
          rv[tid] = rv[tid + s]; ri[tid] = ri[tid + s];
        }
      }
      __syncthreads();
    }
    if (tid == 0) { op[t] = ri[0]; sv[ri[0]] = -3.4e38f; }
    __syncthreads();
  }
}

// ---------------------------------------------------------------- smoothed source: mean of top-14 of 20 neighbor values
__global__ void smooth_kernel(const float* __restrict__ F, int CS, int co, int C,
                              const int* __restrict__ idx, float* __restrict__ src) {
  int i = blockIdx.x * 256 + threadIdx.x;
  if (i >= BB * C * NN) return;
  int n = i % NN, c = (i / NN) % C, b = i / (NN * C);
  const int* ip = idx + ((size_t)b * NN + n) * KNB;
  const float* Fb = F + ((size_t)b * CS + co + c) * NN;
  float vals[KNB]; float tot = 0.f;
#pragma unroll
  for (int t = 0; t < KNB; ++t) { vals[t] = Fb[ip[t]]; tot += vals[t]; }
  unsigned mask = 0;
  for (int r = 0; r < DROP; ++r) {
    float bv = 3.4e38f; int bi = 0;
#pragma unroll
    for (int t = 0; t < KNB; ++t)
      if (!((mask >> t) & 1u) && vals[t] < bv) { bv = vals[t]; bi = t; }
    mask |= 1u << bi; tot -= bv;
  }
  src[((size_t)b * C + c) * NN + n] = tot * (1.f / 14.f);
}

// ---------------------------------------------------------------- f32 weights -> padded f16 [O, Kpad]
__global__ void wcvt_kernel(const float* __restrict__ w, _Float16* __restrict__ o16,
                            int O, int C2, int Kpad) {
  int i = blockIdx.x * 256 + threadIdx.x;
  if (i >= O * Kpad) return;
  int o = i / Kpad, c = i % Kpad;
  o16[i] = (c < C2) ? (_Float16)w[(size_t)o * C2 + c] : (_Float16)0.f;
}

__global__ void zero_kernel(float* p, int n) {
  int i = blockIdx.x * 256 + threadIdx.x;
  if (i < n) p[i] = 0.f;
}

// ---------------------------------------------------------------- edge conv: WMMA f16, fused max over k + BN stats
__global__ void edgeconv_kernel(const float* __restrict__ srcF, int sCS, int sCo,
                                const float* __restrict__ ctrF, int cCS, int cCo,
                                int C, const _Float16* __restrict__ W16, int Kpad, int O,
                                const int* __restrict__ idx,
                                float* __restrict__ maxout,
                                float* __restrict__ sum, float* __restrict__ sumsq) {
  __shared__ _Float16 lE[16 * LDSW];
  int b   = blockIdx.x / NT;
  int n0  = (blockIdx.x % NT) * 16;
  int tid = threadIdx.x;
  int wave = tid >> 5, lane = tid & 31, hi = lane >> 4, l16 = lane & 15;
  int stride = Kpad + 8;
  int C2 = 2 * C;

  v8f  vmax[2];
  float ls[2], lq[2];
#pragma unroll
  for (int s = 0; s < 2; ++s) {
#pragma unroll
    for (int v = 0; v < 8; ++v) vmax[s][v] = -3.4e38f;
    ls[s] = 0.f; lq[s] = 0.f;
  }

  for (int t = 0; t < KNB; ++t) {
    __syncthreads();
    // stage edge features for 16 points at neighbor slot t (f16, padded to Kpad)
    for (int e = tid; e < 16 * Kpad; e += 256) {
      int m = e & 15, c = e >> 4;
      int n = n0 + m;
      float val = 0.f;
      if (c < C) {
        int j = idx[((size_t)b * NN + n) * KNB + t];
        val = srcF[((size_t)b * sCS + sCo + c) * NN + j]
            - ctrF[((size_t)b * cCS + cCo + c) * NN + n];
      } else if (c < C2) {
        val = ctrF[((size_t)b * cCS + cCo + (c - C)) * NN + n];
      }
      lE[m * stride + c] = (_Float16)val;
    }
    __syncthreads();

    int s = 0;
    for (int ot = wave; ot * 16 < O; ot += 8, ++s) {
      v8f acc = {};
      const _Float16* rp = lE + l16 * stride;
      const _Float16* wbase = W16 + ((size_t)(ot * 16 + l16)) * Kpad;
      __builtin_prefetch(wbase, 0, 3);
      for (int k0 = 0; k0 < Kpad; k0 += 32) {
        v16h Af, Bf;
        int koff = k0 + hi * 8;
#pragma unroll
        for (int q = 0; q < 8; ++q) { Af[q] = rp[koff + q]; Af[8 + q] = rp[koff + 16 + q]; }
        const _Float16* wp = wbase + k0 + hi * 8;
#pragma unroll
        for (int q = 0; q < 8; ++q) { Bf[q] = wp[q]; Bf[8 + q] = wp[16 + q]; }
        acc = __builtin_amdgcn_wmma_f32_16x16x32_f16(false, Af, false, Bf,
                                                     (short)0, acc, false, false);
      }
#pragma unroll
      for (int v = 0; v < 8; ++v) {
        float y = acc[v];
        vmax[s][v] = fmaxf(vmax[s][v], y);
        ls[s] += y; lq[s] += y * y;
      }
    }
  }

  int s = 0;
  for (int ot = wave; ot * 16 < O; ot += 8, ++s) {
    int o = ot * 16 + l16;
#pragma unroll
    for (int v = 0; v < 8; ++v)
      maxout[((size_t)b * O + o) * NN + n0 + v + 8 * hi] = vmax[s][v];
    atomicAdd(&sum[o],   ls[s]);
    atomicAdd(&sumsq[o], lq[s]);
  }
}

// ---------------------------------------------------------------- BN(affine)+LeakyReLU applied to the k-max (commutes with max)
__global__ void bnmax_kernel(const float* __restrict__ mx,
                             const float* __restrict__ sum, const float* __restrict__ sumsq,
                             float cnt, float* __restrict__ dest, int dCS, int dCo, int O) {
  int i = blockIdx.x * 256 + threadIdx.x;
  if (i >= BB * O * NN) return;
  int n = i % NN, o = (i / NN) % O, b = i / (NN * O);
  float mu  = sum[o] / cnt;
  float var = sumsq[o] / cnt - mu * mu;
  float y = (mx[((size_t)b * O + o) * NN + n] - mu) * rsqrtf(var + EPSB);
  dest[((size_t)b * dCS + dCo + o) * NN + n] = lrelu(y);
}

// ---------------------------------------------------------------- per-point merge
__global__ void merge_kernel(const int* __restrict__ lidx, const float* __restrict__ xl,
                             const float* __restrict__ xg, float* __restrict__ h) {
  int i = blockIdx.x * 256 + threadIdx.x;
  if (i >= BB * 512 * NN) return;
  int n = i % NN, c = (i / NN) % 512, b = i / (512 * NN);
  size_t o = ((size_t)b * 512 + c) * NN + n;
  h[o] = (lidx[(size_t)b * NN + n] != 0) ? xl[o] : xg[o];
}

// ---------------------------------------------------------------- plain GEMM (w9): async-LDS staged tile, raw y + BN stats
__global__ void gemm_kernel(const float* __restrict__ Fin, int CS, int Cin,
                            const _Float16* __restrict__ W16, int O,
                            float* __restrict__ Y,
                            float* __restrict__ sum, float* __restrict__ sumsq) {
  __shared__ float lHf[512 * S9];       // [c][m] layout, stride 20 floats (16B-aligned rows)
  int b   = blockIdx.x / NT;
  int n0  = (blockIdx.x % NT) * 16;
  int tid = threadIdx.x;
  int wave = tid >> 5, lane = tid & 31, hi = lane >> 4, l16 = lane & 15;

#if ATHENA_ASYNC_LDS
  // async DMA of the 16 x Cin tile into LDS, 16B (b128) granules
  for (int e = tid; e < Cin * 4; e += 256) {
    int c = e >> 2, q = e & 3;
    const float* gp = Fin + ((size_t)b * CS + c) * NN + n0 + q * 4;
    float* lp = lHf + c * S9 + q * 4;
    __builtin_amdgcn_global_load_async_to_lds_b128((GV4*)gp, (LV4*)lp, 0, 0);
  }
  asm volatile("s_wait_asynccnt 0x0" ::: "memory");
#else
  for (int e = tid; e < Cin * 16; e += 256) {
    int c = e >> 4, m = e & 15;
    lHf[c * S9 + m] = Fin[((size_t)b * CS + c) * NN + n0 + m];
  }
#endif
  __syncthreads();

  for (int ot = wave; ot * 16 < O; ot += 8) {
    v8f acc = {};
    const _Float16* wbase = W16 + ((size_t)(ot * 16 + l16)) * Cin;
    __builtin_prefetch(wbase, 0, 3);
    for (int k0 = 0; k0 < Cin; k0 += 32) {
      v16h Af, Bf;
      int koff = k0 + hi * 8;
#pragma unroll
      for (int q = 0; q < 8; ++q) {
        Af[q]     = (_Float16)lHf[(koff + q) * S9 + l16];
        Af[8 + q] = (_Float16)lHf[(koff + 16 + q) * S9 + l16];
      }
      const _Float16* wp = wbase + k0 + hi * 8;
#pragma unroll
      for (int q = 0; q < 8; ++q) { Bf[q] = wp[q]; Bf[8 + q] = wp[16 + q]; }
      acc = __builtin_amdgcn_wmma_f32_16x16x32_f16(false, Af, false, Bf,
                                                   (short)0, acc, false, false);
    }
    int o = ot * 16 + l16;
    float lsv = 0.f, lqv = 0.f;
#pragma unroll
    for (int v = 0; v < 8; ++v) {
      float y = acc[v];
      Y[((size_t)b * O + o) * NN + n0 + v + 8 * hi] = y;
      lsv += y; lqv += y * y;
    }
    atomicAdd(&sum[o], lsv);
    atomicAdd(&sumsq[o], lqv);
  }
}

// ---------------------------------------------------------------- max & mean pool over N with BN+act on the fly
__global__ void pool_kernel(const float* __restrict__ Y,
                            const float* __restrict__ sum, const float* __restrict__ sumsq,
                            float cnt, float* __restrict__ g) {
  __shared__ float rm[256];
  __shared__ float rs[256];
  int tid = threadIdx.x;
  int b = blockIdx.x / 1024, o = blockIdx.x % 1024;
  float mu  = sum[o] / cnt;
  float var = sumsq[o] / cnt - mu * mu;
  float rstd = rsqrtf(var + EPSB);
  const float* yp = Y + ((size_t)b * 1024 + o) * NN;
  float lmax = -3.4e38f, lsum = 0.f;
  for (int n = tid; n < NN; n += 256) {
    float y = lrelu((yp[n] - mu) * rstd);
    lmax = fmaxf(lmax, y); lsum += y;
  }
  rm[tid] = lmax; rs[tid] = lsum; __syncthreads();
  for (int s = 128; s > 0; s >>= 1) {
    if (tid < s) { rm[tid] = fmaxf(rm[tid], rm[tid + s]); rs[tid] += rs[tid + s]; }
    __syncthreads();
  }
  if (tid == 0) {
    g[(size_t)b * 2048 + o]        = rm[0];
    g[(size_t)b * 2048 + 1024 + o] = rs[0] / (float)NN;
  }
}

// ---------------------------------------------------------------- MLP head (batch of 4) in one block
__global__ void head_kernel(const float* __restrict__ g,
                            const float* __restrict__ l1w,
                            const float* __restrict__ l2w, const float* __restrict__ l2b,
                            const float* __restrict__ l3w, const float* __restrict__ l3b,
                            float* __restrict__ out) {
  __shared__ float h1[4][512];
  __shared__ float h2[4][256];
  int tid = threadIdx.x;
  if (tid < 512) {
    for (int bb = 0; bb < 4; ++bb) {
      float a = 0.f;
      const float* gp = g + (size_t)bb * 2048;
      const float* wp = l1w + (size_t)tid * 2048;
      for (int c = 0; c < 2048; ++c) a += gp[c] * wp[c];
      h1[bb][tid] = a;
    }
  }
  __syncthreads();
  if (tid < 512) {
    float mu = 0.f;
    for (int bb = 0; bb < 4; ++bb) mu += h1[bb][tid];
    mu *= 0.25f;
    float var = 0.f;
    for (int bb = 0; bb < 4; ++bb) { float d = h1[bb][tid] - mu; var += d * d; }
    var *= 0.25f;
    float rs = rsqrtf(var + EPSB);
    for (int bb = 0; bb < 4; ++bb) h1[bb][tid] = lrelu((h1[bb][tid] - mu) * rs);
  }
  __syncthreads();
  if (tid < 256) {
    for (int bb = 0; bb < 4; ++bb) {
      float a = l2b[tid];
      const float* wp = l2w + (size_t)tid * 512;
      for (int c = 0; c < 512; ++c) a += h1[bb][c] * wp[c];
      h2[bb][tid] = a;
    }
  }
  __syncthreads();
  if (tid < 256) {
    float mu = 0.f;
    for (int bb = 0; bb < 4; ++bb) mu += h2[bb][tid];
    mu *= 0.25f;
    float var = 0.f;
    for (int bb = 0; bb < 4; ++bb) { float d = h2[bb][tid] - mu; var += d * d; }
    var *= 0.25f;
    float rs = rsqrtf(var + EPSB);
    for (int bb = 0; bb < 4; ++bb) h2[bb][tid] = lrelu((h2[bb][tid] - mu) * rs);
  }
  __syncthreads();
  if (tid < 40) {
    for (int bb = 0; bb < 4; ++bb) {
      float a = l3b[tid];
      const float* wp = l3w + (size_t)tid * 256;
      for (int c = 0; c < 256; ++c) a += h2[bb][c] * wp[c];
      out[bb * 40 + tid] = a;
    }
  }
}

// ================================================================ host orchestration
extern "C" void kernel_launch(void* const* d_in, const int* in_sizes, int n_in,
                              void* d_out, int out_size, void* d_ws, size_t ws_size,
                              hipStream_t stream) {
  const float* x    = (const float*)d_in[0];
  const int*   lidx = (const int*)d_in[1];
  const float* w[9];
  for (int i = 0; i < 9; ++i) w[i] = (const float*)d_in[3 + i];
  const float* l1w = (const float*)d_in[12];
  const float* l2w = (const float*)d_in[13];
  const float* l2b = (const float*)d_in[14];
  const float* l3w = (const float*)d_in[15];
  const float* l3b = (const float*)d_in[16];

  size_t off = 0;
  auto carve = [&](size_t bytes) -> char* {
    char* p = (char*)d_ws + off;
    off += (bytes + 255) & ~(size_t)255;
    return p;
  };
  float*    pd   = (float*)carve((size_t)BB * NN * NN * 4);
  int*      idx  = (int*)carve((size_t)BB * NN * KNB * 4);
  float*    xx   = (float*)carve((size_t)BB * NN * 4);
  float*    xpad = (float*)carve((size_t)BB * 4 * NN * 4);
  float*    xloc = (float*)carve((size_t)BB * 512 * NN * 4);
  float*    xglb = (float*)carve((size_t)BB * 512 * NN * 4);
  float*    srcb = (float*)carve((size_t)BB * 256 * NN * 4);
  float*    mx   = (float*)carve((size_t)BB * 256 * NN * 4);
  _Float16* w16  = (_Float16*)carve((size_t)1024 * 512 * 2);
  float*    stat = (float*)carve((size_t)2048 * 4);          // sum | sumsq
  float*    h    = (float*)carve((size_t)BB * 512 * NN * 4);
  float*    y9   = (float*)carve((size_t)BB * 1024 * NN * 4);
  float*    g    = (float*)carve((size_t)BB * 2048 * 4);

  pad_kernel<<<(BB * 4 * NN + 255) / 256, 256, 0, stream>>>(x, xpad);

  struct L { int co, C, O, ch; };
  const L Ls[4] = { {0, 3, 64, 0}, {0, 64, 64, 64}, {64, 64, 128, 128}, {128, 128, 256, 256} };

  for (int br = 0; br < 2; ++br) {            // 0 = local (smoothed), 1 = global
    float* Xb = br ? xglb : xloc;
    int invert = br;                          // keep = local_idx for local, !local_idx for global
    for (int li = 0; li < 4; ++li) {
      const float* F = (li == 0) ? xpad : Xb;
      int CS = (li == 0) ? 4 : 512;
      int co = Ls[li].co, C = Ls[li].C, O = Ls[li].O, ch = Ls[li].ch;
      int Cd = (li == 0) ? 4 : C;             // dist/xx channel count (padded, %4==0)
      int C2 = 2 * C;
      int Kpad = ((C2 + 31) / 32) * 32;
      const float* wl = w[br * 4 + li];

      xx_kernel<<<(BB * NN + 255) / 256, 256, 0, stream>>>(F, CS, co, Cd, xx);
      dist_kernel<<<(BB * NT * (NT / JT)) / 8, 256, 0, stream>>>(F, CS, co, Cd, xx, pd);
      topk_kernel<<<BB * NN, 256, 0, stream>>>(pd, lidx, invert, idx);

      const float* srcF = F; int sCS = CS, sCo = co;
      if (br == 0) {
        smooth_kernel<<<(BB * C * NN + 255) / 256, 256, 0, stream>>>(F, CS, co, C, idx, srcb);
        srcF = srcb; sCS = C; sCo = 0;
      }
      zero_kernel<<<8, 256, 0, stream>>>(stat, 2048);
      wcvt_kernel<<<(O * Kpad + 255) / 256, 256, 0, stream>>>(wl, w16, O, C2, Kpad);
      edgeconv_kernel<<<BB * NT, 256, 0, stream>>>(srcF, sCS, sCo, F, CS, co, C,
                                                   w16, Kpad, O, idx,
                                                   mx, stat, stat + 1024);
      bnmax_kernel<<<(BB * O * NN + 255) / 256, 256, 0, stream>>>(
          mx, stat, stat + 1024, (float)(BB * NN * KNB), Xb, 512, ch, O);
    }
  }

  merge_kernel<<<(BB * 512 * NN + 255) / 256, 256, 0, stream>>>(lidx, xloc, xglb, h);

  zero_kernel<<<8, 256, 0, stream>>>(stat, 2048);
  wcvt_kernel<<<(1024 * 512 + 255) / 256, 256, 0, stream>>>(w[8], w16, 1024, 512, 512);
  gemm_kernel<<<BB * NT, 256, 0, stream>>>(h, 512, 512, w16, 1024, y9, stat, stat + 1024);
  pool_kernel<<<BB * 1024, 256, 0, stream>>>(y9, stat, stat + 1024, (float)(BB * NN), g);
  head_kernel<<<1, 512, 0, stream>>>(g, l1w, l2w, l2b, l3w, l3b, (float*)d_out);
}
